// HeteroGraphSAGE_52931176955955
// MI455X (gfx1250) — compile-verified, matrix operand order checked by reading, not run on hardware
//
#include <hip/hip_runtime.h>

typedef __attribute__((ext_vector_type(16))) _Float16 v16h;
typedef __attribute__((ext_vector_type(8)))  float    v8f;

#define DD 128
#define WELEMS (DD * DD)     // 16384 per matrix
#define LDSS 136             // LDS row stride in halves (bank-conflict-free, 16B aligned)

// ---------------------------------------------------------------------------
// Weight prep: W[mat][k][n] fp32 -> transposed [mat][n][k] split into f16 hi/lo
// ---------------------------------------------------------------------------
__global__ void prep_weights(const float* __restrict__ Wl, const float* __restrict__ Wr,
                             _Float16* __restrict__ lhi, _Float16* __restrict__ llo,
                             _Float16* __restrict__ rhi, _Float16* __restrict__ rlo)
{
    int idx = blockIdx.x * blockDim.x + threadIdx.x;
    if (idx >= 11 * WELEMS) return;
    int mat = idx >> 14;
    int rem = idx & (WELEMS - 1);
    int nn  = rem >> 7;
    int kk  = rem & (DD - 1);
    size_t sidx = ((size_t)mat << 14) + (size_t)kk * DD + nn;

    float v = Wl[sidx];
    _Float16 h = (_Float16)v;
    lhi[idx] = h;
    llo[idx] = (_Float16)(v - (float)h);

    v = Wr[sidx];
    h = (_Float16)v;
    rhi[idx] = h;
    rlo[idx] = (_Float16)(v - (float)h);
}

// ---------------------------------------------------------------------------
// Edge scatter: one wave per edge, float4 per lane (128 floats), float atomics
// lane 0 also bumps the destination degree count.
// ---------------------------------------------------------------------------
__global__ void scatter_add(const int* __restrict__ src, const int* __restrict__ dst, int E,
                            const float* __restrict__ x, float* __restrict__ agg,
                            float* __restrict__ cnt)
{
    int g = blockIdx.x * blockDim.x + threadIdx.x;
    int e = g >> 5;
    int lane = g & 31;
    if (e >= E) return;
    int s = src[e];
    int d = dst[e];
    float4 v = ((const float4*)(x + (size_t)s * DD))[lane];
    float* base = agg + (size_t)d * DD + lane * 4;
    atomicAdd(base + 0, v.x);
    atomicAdd(base + 1, v.y);
    atomicAdd(base + 2, v.z);
    atomicAdd(base + 3, v.w);
    if (lane == 0) atomicAdd(cnt + d, 1.0f);
}

// ---------------------------------------------------------------------------
// Fragment gather: 16 halves for one A-row / B-column of a 16x16x32 f16 WMMA.
// half h in [0,8): K = kbase + h ; h in [8,16): K = kbase + 16 + (h-8)
// (kbase = 32*chunk + 8*laneHi) -- matches the ISA 16-bit A/B VGPR layout.
// ---------------------------------------------------------------------------
__device__ __forceinline__ v16h load_frag16(const _Float16* p, int kbase)
{
    v16h r;
#pragma unroll
    for (int i = 0; i < 8; ++i) {
        r[i]     = p[kbase + i];
        r[i + 8] = p[kbase + 16 + i];
    }
    return r;
}

// ---------------------------------------------------------------------------
// Fused SAGE transform:
//   out[r,:] = (agg[r,:]/max(cnt[r],1)) @ Wl + bias (+ root[r,:] @ Wr) (+= out)
// 256 threads = 8 wave32; block handles 16 rows; wave w handles cols [16w,16w+16).
// f16 hi/lo split => 3 WMMAs per K-chunk per weight matrix, f32 accumulation.
// ---------------------------------------------------------------------------
__global__ __launch_bounds__(256)
void sage_gemm(const float* __restrict__ agg, const float* __restrict__ cnt,
               const float* __restrict__ root,
               const _Float16* __restrict__ lhi, const _Float16* __restrict__ llo,
               const _Float16* __restrict__ rhi, const _Float16* __restrict__ rlo,
               const float* __restrict__ bias,
               float* __restrict__ out, int n, int accum)
{
    __shared__ _Float16 sAhi[16 * LDSS], sAlo[16 * LDSS];
    __shared__ _Float16 sRhi[16 * LDSS], sRlo[16 * LDSS];

    const int tid  = threadIdx.x;
    const int row0 = blockIdx.x * 16;
    const bool has_root = (root != nullptr);

    // Stage A (mean-divided) and root rows into LDS as f16 hi/lo.
    for (int idx = tid; idx < 16 * DD; idx += 256) {
        int r = idx >> 7, k = idx & (DD - 1);
        int grow = row0 + r;
        float v = 0.f;
        if (grow < n) {
            float c = cnt[grow];
            v = agg[(size_t)grow * DD + k] * (1.0f / fmaxf(c, 1.0f));
        }
        _Float16 h = (_Float16)v;
        sAhi[r * LDSS + k] = h;
        sAlo[r * LDSS + k] = (_Float16)(v - (float)h);
        if (has_root) {
            float rv = (grow < n) ? root[(size_t)grow * DD + k] : 0.f;
            _Float16 rh = (_Float16)rv;
            sRhi[r * LDSS + k] = rh;
            sRlo[r * LDSS + k] = (_Float16)(rv - (float)rh);
        }
    }
    __syncthreads();

    const int wave = tid >> 5;
    const int lane = tid & 31;
    const int lh   = lane >> 4;   // lane-half selects K sub-block per ISA layout
    const int mn   = lane & 15;   // A row (M) / B column (N) within the tile
    const int n0   = wave * 16;

    const _Float16* aHi = sAhi + mn * LDSS;
    const _Float16* aLo = sAlo + mn * LDSS;
    const _Float16* bHi = lhi + (size_t)(n0 + mn) * DD;
    const _Float16* bLo = llo + (size_t)(n0 + mn) * DD;

    v8f acc = {};
#pragma unroll
    for (int kc = 0; kc < 4; ++kc) {
        const int kbase = kc * 32 + lh * 8;
        v16h ah = load_frag16(aHi, kbase);
        v16h al = load_frag16(aLo, kbase);
        v16h bh = load_frag16(bHi, kbase);
        v16h bl = load_frag16(bLo, kbase);
        acc = __builtin_amdgcn_wmma_f32_16x16x32_f16(false, ah, false, bh, (short)0, acc, false, false);
        acc = __builtin_amdgcn_wmma_f32_16x16x32_f16(false, ah, false, bl, (short)0, acc, false, false);
        acc = __builtin_amdgcn_wmma_f32_16x16x32_f16(false, al, false, bh, (short)0, acc, false, false);
    }
    if (has_root) {
        const _Float16* cHi = sRhi + mn * LDSS;
        const _Float16* cLo = sRlo + mn * LDSS;
        const _Float16* dHi = rhi + (size_t)(n0 + mn) * DD;
        const _Float16* dLo = rlo + (size_t)(n0 + mn) * DD;
#pragma unroll
        for (int kc = 0; kc < 4; ++kc) {
            const int kbase = kc * 32 + lh * 8;
            v16h ah = load_frag16(cHi, kbase);
            v16h al = load_frag16(cLo, kbase);
            v16h bh = load_frag16(dHi, kbase);
            v16h bl = load_frag16(dLo, kbase);
            acc = __builtin_amdgcn_wmma_f32_16x16x32_f16(false, ah, false, bh, (short)0, acc, false, false);
            acc = __builtin_amdgcn_wmma_f32_16x16x32_f16(false, ah, false, bl, (short)0, acc, false, false);
            acc = __builtin_amdgcn_wmma_f32_16x16x32_f16(false, al, false, bh, (short)0, acc, false, false);
        }
    }

    // C/D layout: VGPR r2, lane -> (M = r2 + 8*laneHi, N = lane&15)
    const float bv = bias[n0 + mn];
#pragma unroll
    for (int r2 = 0; r2 < 8; ++r2) {
        int grow = row0 + r2 + lh * 8;
        if (grow < n) {
            size_t o = (size_t)grow * DD + n0 + mn;
            float v = acc[r2] + bv;
            out[o] = accum ? (out[o] + v) : v;
        }
    }
}

// ---------------------------------------------------------------------------
// BatchNorm (training stats) with fused ReLU: pass 1 = per-column sum/sumsq.
// ---------------------------------------------------------------------------
__global__ void bn_stats(const float* __restrict__ x, int n, float* __restrict__ stats)
{
    __shared__ float ssum[DD], ssq[DD];
    int tid = threadIdx.x;
    if (tid < DD) { ssum[tid] = 0.f; ssq[tid] = 0.f; }
    __syncthreads();
    int c = tid & (DD - 1);
    int half = tid >> 7;
    float s = 0.f, q = 0.f;
    for (int r = blockIdx.x * 2 + half; r < n; r += gridDim.x * 2) {
        float v = fmaxf(x[(size_t)r * DD + c], 0.f);
        s += v;
        q += v * v;
    }
    atomicAdd(&ssum[c], s);
    atomicAdd(&ssq[c], q);
    __syncthreads();
    if (tid < DD) {
        atomicAdd(&stats[tid], ssum[tid]);
        atomicAdd(&stats[DD + tid], ssq[tid]);
    }
}

__global__ void bn_apply(float* __restrict__ x, int n, const float* __restrict__ stats,
                         const float* __restrict__ g, const float* __restrict__ b)
{
    size_t idx = (size_t)blockIdx.x * blockDim.x + threadIdx.x;
    size_t tot = (size_t)n * DD;
    if (idx >= tot) return;
    int c = (int)(idx & (DD - 1));
    float inv_n = 1.0f / (float)n;
    float m   = stats[c] * inv_n;
    float var = stats[DD + c] * inv_n - m * m;
    float v = fmaxf(x[idx], 0.f);
    x[idx] = (v - m) * rsqrtf(var + 1e-5f) * g[c] + b[c];
}

// ---------------------------------------------------------------------------
// Host-side orchestration helpers
// ---------------------------------------------------------------------------
static void run_conv(const int* e, int E, const float* src_feats, const float* root_feats,
                     int n_dst, int widx,
                     const _Float16* lhi, const _Float16* llo,
                     const _Float16* rhi, const _Float16* rlo,
                     const float* bL, float* agg, float* cnt,
                     float* out, int accum, hipStream_t s)
{
    hipMemsetAsync(agg, 0, (size_t)n_dst * DD * sizeof(float), s);
    hipMemsetAsync(cnt, 0, (size_t)n_dst * sizeof(float), s);
    long total = (long)E * 32;
    scatter_add<<<(unsigned)((total + 255) / 256), 256, 0, s>>>(e, e + E, E, src_feats, agg, cnt);
    dim3 grid((n_dst + 15) / 16);
    sage_gemm<<<grid, 256, 0, s>>>(agg, cnt, root_feats,
        lhi + (size_t)widx * WELEMS, llo + (size_t)widx * WELEMS,
        root_feats ? rhi + (size_t)widx * WELEMS : nullptr,
        root_feats ? rlo + (size_t)widx * WELEMS : nullptr,
        bL + widx * DD, out, n_dst, accum);
}

static void run_bn(float* x, int n, int gidx, const float* bn_g, const float* bn_b,
                   float* stats, hipStream_t s)
{
    hipMemsetAsync(stats, 0, 2 * DD * sizeof(float), s);
    bn_stats<<<512, 256, 0, s>>>(x, n, stats);
    size_t tot = (size_t)n * DD;
    bn_apply<<<(unsigned)((tot + 255) / 256), 256, 0, s>>>(x, n, stats,
                                                           bn_g + gidx * DD, bn_b + gidx * DD);
}

extern "C" void kernel_launch(void* const* d_in, const int* in_sizes, int n_in,
                              void* d_out, int out_size, void* d_ws, size_t ws_size,
                              hipStream_t stream)
{
    const float* x_paper  = (const float*)d_in[0];
    const float* x_author = (const float*)d_in[1];
    const float* x_inst   = (const float*)d_in[2];
    const int*   e_pp     = (const int*)d_in[3];
    const int*   e_pa     = (const int*)d_in[4];
    const int*   e_ap     = (const int*)d_in[5];
    const int*   e_ai     = (const int*)d_in[6];
    const int*   e_ia     = (const int*)d_in[7];
    const float* Wl       = (const float*)d_in[8];
    const float* bL       = (const float*)d_in[9];
    const float* Wr       = (const float*)d_in[10];
    const float* bn_g     = (const float*)d_in[11];
    const float* bn_b     = (const float*)d_in[12];

    const int NP = in_sizes[0] / DD;
    const int NA = in_sizes[1] / DD;
    const int NI = in_sizes[2] / DD;
    const int Epp = in_sizes[3] / 2;
    const int Epa = in_sizes[4] / 2;
    const int Eap = in_sizes[5] / 2;
    const int Eai = in_sizes[6] / 2;
    const int Eia = in_sizes[7] / 2;

    // Workspace carve-up (256B aligned)
    char* ws = (char*)d_ws;
    size_t off = 0;
    auto carve = [&](size_t bytes) -> void* {
        void* p = ws + off;
        off = (off + bytes + 255) & ~(size_t)255;
        return p;
    };
    _Float16* WlHi = (_Float16*)carve((size_t)11 * WELEMS * sizeof(_Float16));
    _Float16* WlLo = (_Float16*)carve((size_t)11 * WELEMS * sizeof(_Float16));
    _Float16* WrHi = (_Float16*)carve((size_t)11 * WELEMS * sizeof(_Float16));
    _Float16* WrLo = (_Float16*)carve((size_t)11 * WELEMS * sizeof(_Float16));
    float* agg   = (float*)carve((size_t)NP * DD * sizeof(float));
    float* cnt   = (float*)carve((size_t)NP * sizeof(float));
    float* p0    = (float*)carve((size_t)NP * DD * sizeof(float));
    float* a0    = (float*)carve((size_t)NA * DD * sizeof(float));
    float* p1    = (float*)carve((size_t)NP * DD * sizeof(float));
    float* a1    = (float*)carve((size_t)NA * DD * sizeof(float));
    float* i1    = (float*)carve((size_t)NI * DD * sizeof(float));
    float* stats = (float*)carve(2 * DD * sizeof(float));
    (void)ws_size; (void)n_in; (void)out_size;

    float* p2 = (float*)d_out;
    float* a2 = p2 + (size_t)NP * DD;
    float* i2 = a2 + (size_t)NA * DD;

    // One-time weight split/transpose (deterministic, re-run every call)
    prep_weights<<<(11 * WELEMS + 255) / 256, 256, 0, stream>>>(Wl, Wr, WlHi, WlLo, WrHi, WrLo);

    // ---- Layer 0 ----
    run_conv(e_pp, Epp, x_paper, x_paper, NP, 0, WlHi, WlLo, WrHi, WrLo, bL, agg, cnt, p0, 0, stream);
    run_bn(p0, NP, 0, bn_g, bn_b, stats, stream);
    run_conv(e_pa, Epa, x_paper, x_author, NA, 1, WlHi, WlLo, WrHi, WrLo, bL, agg, cnt, a0, 0, stream);
    run_bn(a0, NA, 1, bn_g, bn_b, stats, stream);

    // ---- Layer 1 ----
    run_conv(e_pp, Epp, p0, p0, NP, 2, WlHi, WlLo, WrHi, WrLo, bL, agg, cnt, p1, 0, stream);
    run_conv(e_ap, Eap, a0, p0, NP, 4, WlHi, WlLo, WrHi, WrLo, bL, agg, cnt, p1, 1, stream);
    run_bn(p1, NP, 2, bn_g, bn_b, stats, stream);
    run_conv(e_pa, Epa, p0, a0, NA, 3, WlHi, WlLo, WrHi, WrLo, bL, agg, cnt, a1, 0, stream);
    run_bn(a1, NA, 3, bn_g, bn_b, stats, stream);
    run_conv(e_ai, Eai, a0, nullptr, NI, 5, WlHi, WlLo, WrHi, WrLo, bL, agg, cnt, i1, 0, stream);
    run_bn(i1, NI, 4, bn_g, bn_b, stats, stream);

    // ---- Layer 2 (outputs, no relu/BN) ----
    run_conv(e_pp, Epp, p1, p1, NP, 6,  WlHi, WlLo, WrHi, WrLo, bL, agg, cnt, p2, 0, stream);
    run_conv(e_ap, Eap, a1, p1, NP, 8,  WlHi, WlLo, WrHi, WrLo, bL, agg, cnt, p2, 1, stream);
    run_conv(e_pa, Epa, p1, a1, NA, 7,  WlHi, WlLo, WrHi, WrLo, bL, agg, cnt, a2, 0, stream);
    run_conv(e_ia, Eia, i1, a1, NA, 10, WlHi, WlLo, WrHi, WrLo, bL, agg, cnt, a2, 1, stream);
    run_conv(e_ai, Eai, a1, i1, NI, 9,  WlHi, WlLo, WrHi, WrLo, bL, agg, cnt, i2, 0, stream);
}
